// GraphTransformer2_2164663517448
// MI455X (gfx1250) — compile-verified, matrix operand order checked by reading, not run on hardware
//
#include <hip/hip_runtime.h>

// ---------------------------------------------------------------------------
// CDNA5 (gfx1250) GraphTransformer forward. bf16 WMMA everywhere, fp32 accum.
// Weights pre-transposed in global so all WMMA fragments read contiguous-K;
// weight tiles staged to LDS with DOUBLE-BUFFERED global_load_async_to_lds_b128
// (ASYNCcnt), overlapping copy latency with WMMA math.
// ---------------------------------------------------------------------------

typedef __attribute__((ext_vector_type(16))) __bf16 v16bf;
typedef __attribute__((ext_vector_type(8)))  float  v8f;

__device__ __forceinline__ v8f wmma_bf16(v16bf a, v16bf b, v8f c) {
  // D = A(16x32 bf16) * B(32x16 bf16) + C(16x16 f32)
  return __builtin_amdgcn_wmma_f32_16x16x32_bf16(
      /*neg_a=*/false, a, /*neg_b=*/false, b,
      /*c_mod=*/(short)0, c, /*reuse_a=*/false, /*reuse_b=*/false);
}

// 16x32 bf16 A-style fragment (valid for B fragments by symmetry: lane%16
// selects row(m)/col(n), lane/16 selects K-half, VGPRs hold K pairs).
// elements 0..7  = base[row*stride + k0 + (hi?8:0) + i]
// elements 8..15 = same + 16   (contiguous -> b128/paired-dword loads)
__device__ __forceinline__ v16bf load_frag(const __bf16* __restrict__ base,
                                           int stride, int row, int k0, int hi) {
  const __bf16* p = base + (size_t)row * stride + k0 + (hi ? 8 : 0);
  v16bf a;
#pragma unroll
  for (int i = 0; i < 8; ++i) a[i] = p[i];
#pragma unroll
  for (int i = 0; i < 8; ++i) a[8 + i] = p[16 + i];
  return a;
}

__device__ __forceinline__ float gelu_f(float x) {
  float x3 = x * x * x;
  return 0.5f * x * (1.0f + tanhf(0.7978845608028654f * (x + 0.044715f * x3)));
}

__device__ __forceinline__ float rmax16(float v) {
  v = fmaxf(v, __shfl_xor(v, 1, 32));
  v = fmaxf(v, __shfl_xor(v, 2, 32));
  v = fmaxf(v, __shfl_xor(v, 4, 32));
  v = fmaxf(v, __shfl_xor(v, 8, 32));
  return v;
}
__device__ __forceinline__ float rsum16(float v) {
  v += __shfl_xor(v, 1, 32);
  v += __shfl_xor(v, 2, 32);
  v += __shfl_xor(v, 4, 32);
  v += __shfl_xor(v, 8, 32);
  return v;
}

// ---------------------------------------------------------------------------
// GEMM: C[M,N] = A[M,K] @ Bt^T + bias, optional GELU.
// A bf16 row-major [M][K]; Bt bf16 PRE-TRANSPOSED [N][K].
// Block 256 thr (8 waves, 4Mx2N), block tile 128x128, wave tile 32x64,
// k-step 32.  Weight tiles double-buffered in LDS via async global->LDS
// copies: tile t+1 is issued before tile t is consumed; s_wait_asynccnt 2
// leaves only the prefetched batch in flight.
// ---------------------------------------------------------------------------
#define GBM 128
#define GBN 128
#define GBK 32
#define BSTR 40  // LDS row stride in halves: 80B, keeps 16B-aligned chunks

__global__ void __launch_bounds__(256)
gemm_bf16_kernel(const __bf16* __restrict__ A, const __bf16* __restrict__ Bt,
                 const float* __restrict__ bias,
                 float* __restrict__ outF, __bf16* __restrict__ outB,
                 __bf16* __restrict__ outBT,
                 int M, int N, int K, int doGelu) {
  __shared__ __bf16 Bl[2][GBN][BSTR];  // double-buffered [n_local][k_local]

  const int tid = threadIdx.x;
  const int lane = tid & 31;
  const int l16 = lane & 15;
  const int hi = (lane >> 4) & 1;
  const int wave = tid >> 5;
  const int wm = (wave & 3) * 32;   // wave M offset in block tile
  const int wn = (wave >> 2) * 64;  // wave N offset in block tile
  const int mbase = blockIdx.y * GBM;
  const int nbase = blockIdx.x * GBN;

  v8f acc[2][4] = {};

  const unsigned lds0 = (unsigned)(uintptr_t)&Bl[0][0][0];
  const unsigned lds1 = (unsigned)(uintptr_t)&Bl[1][0][0];

  // each thread copies 2 x 16B chunks of the 128x32 weight tile
  auto issue_tile = [&](int k0, unsigned lbase) {
#pragma unroll
    for (int j = 0; j < 2; ++j) {
      const int c = tid * 2 + j;      // 0..511 chunks of 16B
      const int n = c >> 2;           // 0..127
      const int kq = (c & 3) * 8;     // 0,8,16,24
      const __bf16* g = Bt + (size_t)(nbase + n) * K + k0 + kq;
      const unsigned laddr = lbase + (unsigned)(n * BSTR + kq) * 2u;
      asm volatile("global_load_async_to_lds_b128 %0, %1, off"
                   :: "v"(laddr), "v"((unsigned long long)(uintptr_t)g)
                   : "memory");
    }
  };

  issue_tile(0, lds0);
  const int nsteps = K / GBK;

  for (int t = 0; t < nsteps; ++t) {
    const int k0 = t * GBK;
    const __bf16* buf = &Bl[t & 1][0][0];
    if (t + 1 < nsteps) {
      issue_tile(k0 + GBK, (t & 1) ? lds0 : lds1);  // prefetch next tile
      asm volatile("s_wait_asynccnt 0x2" ::: "memory");  // current tile landed
    } else {
      asm volatile("s_wait_asynccnt 0x0" ::: "memory");
    }
    __syncthreads();

    const v16bf a0 = load_frag(A, K, mbase + wm + 0  + l16, k0, hi);
    const v16bf a1 = load_frag(A, K, mbase + wm + 16 + l16, k0, hi);
#pragma unroll
    for (int fn = 0; fn < 4; ++fn) {
      const v16bf b = load_frag(buf, BSTR, wn + fn * 16 + l16, 0, hi);
      acc[0][fn] = wmma_bf16(a0, b, acc[0][fn]);
      acc[1][fn] = wmma_bf16(a1, b, acc[1][fn]);
    }
    __syncthreads();  // all waves done with buf before it is re-filled at t+2
  }

  // ---- epilogue ----
#pragma unroll
  for (int fm = 0; fm < 2; ++fm) {
#pragma unroll
    for (int fn = 0; fn < 4; ++fn) {
      const int n = nbase + wn + fn * 16 + l16;
      const float bv = bias ? bias[n] : 0.0f;
#pragma unroll
      for (int r = 0; r < 8; ++r) {
        const int m = mbase + wm + fm * 16 + r + 8 * hi;
        float v = acc[fm][fn][r] + bv;
        if (doGelu) v = gelu_f(v);
        if (outF)  outF[(size_t)m * N + n] = v;
        if (outB)  outB[(size_t)m * N + n] = (__bf16)v;
        if (outBT) outBT[(size_t)n * M + m] = (__bf16)v;  // transposed store (Vt)
      }
    }
  }
}

// ---------------------------------------------------------------------------
// Masked multi-head attention, flash-style online softmax.
// Block = 128 threads = 4 waves = 4 heads; one 16-row query block per block.
// Per 32-key step: S(16x32) via 4 WMMA, mask+scale, online softmax with
// shfl_xor row reductions, P transposed through PER-WAVE LDS (no block
// barriers; LDS is in-order per wave, enforced with s_wait_dscnt), PV via
// 4 WMMA.  Q,K token-major bf16 (ld=256); V transposed (feature-major, ld=sz).
// adj (the dominant HBM stream) is prefetched one key-tile ahead.
// ---------------------------------------------------------------------------
__global__ void __launch_bounds__(128)
attn_kernel(const __bf16* __restrict__ Q, const __bf16* __restrict__ Km,
            const __bf16* __restrict__ Vt, const int* __restrict__ adj,
            __bf16* __restrict__ outB, int sz) {
  __shared__ __bf16 pbuf[4][16][36];  // per-wave slices, no cross-wave sharing

  const int tid = threadIdx.x;
  const int lane = tid & 31;
  const int l16 = lane & 15;
  const int hi = (lane >> 4) & 1;
  const int h = tid >> 5;         // head 0..3
  const int hoff = h * 64;
  const int qbase = blockIdx.x * 16;
  const float scale = 0.125f;     // 1/sqrt(dh=64)

  const v16bf qa0 = load_frag(Q, 256, qbase + l16, hoff + 0, hi);
  const v16bf qa1 = load_frag(Q, 256, qbase + l16, hoff + 32, hi);

  float mrow[8], lrow[8];
#pragma unroll
  for (int r = 0; r < 8; ++r) { mrow[r] = -3.0e38f; lrow[r] = 0.0f; }
  v8f o[4] = {};

  for (int kt = 0; kt < sz; kt += 32) {
    // prefetch next tile's mask rows (lanes 0..15 cover the 16 query rows)
    if (kt + 32 < sz) {
      __builtin_prefetch(adj + (size_t)(qbase + l16) * sz + kt + 32, 0, 0);
    }

    // ---- S = Q @ K^T for two 16-key subtiles ----
    v8f S0 = {}, S1 = {};
    {
      const v16bf bk0a = load_frag(Km, 256, kt + l16, hoff + 0, hi);
      const v16bf bk0b = load_frag(Km, 256, kt + l16, hoff + 32, hi);
      S0 = wmma_bf16(qa0, bk0a, S0);
      S0 = wmma_bf16(qa1, bk0b, S0);
      const v16bf bk1a = load_frag(Km, 256, kt + 16 + l16, hoff + 0, hi);
      const v16bf bk1b = load_frag(Km, 256, kt + 16 + l16, hoff + 32, hi);
      S1 = wmma_bf16(qa0, bk1a, S1);
      S1 = wmma_bf16(qa1, bk1b, S1);
    }

    // ---- mask + scale + online softmax ----
    float s0[8], s1[8];
#pragma unroll
    for (int r = 0; r < 8; ++r) {
      const int q = qbase + r + 8 * hi;
      const size_t base = (size_t)q * sz + kt;
      s0[r] = (adj[base + l16] == 0)      ? -1.0e9f : S0[r] * scale;
      s1[r] = (adj[base + 16 + l16] == 0) ? -1.0e9f : S1[r] * scale;
    }
#pragma unroll
    for (int r = 0; r < 8; ++r) {
      float t = rmax16(fmaxf(s0[r], s1[r]));
      const float mnew = fmaxf(mrow[r], t);
      const float alpha = __expf(mrow[r] - mnew);
      mrow[r] = mnew;
      const float p0 = __expf(s0[r] - mnew);
      const float p1 = __expf(s1[r] - mnew);
      lrow[r] = lrow[r] * alpha + rsum16(p0 + p1);
#pragma unroll
      for (int a = 0; a < 4; ++a) o[a][r] *= alpha;
      pbuf[h][r + 8 * hi][l16]      = (__bf16)p0;
      pbuf[h][r + 8 * hi][16 + l16] = (__bf16)p1;
    }
    // per-wave ordering: all ds stores complete before fragment reload
    asm volatile("s_wait_dscnt 0x0" ::: "memory");

    // ---- O += P(16x32) @ V(32x64) ----
    const v16bf pa = load_frag(&pbuf[h][0][0], 36, l16, 0, hi);
#pragma unroll
    for (int fn = 0; fn < 4; ++fn) {
      const v16bf bv = load_frag(Vt, sz, hoff + fn * 16 + l16, kt, hi);
      o[fn] = wmma_bf16(pa, bv, o[fn]);
    }
    asm volatile("" ::: "memory");  // keep next iteration's stores below loads
  }

  // ---- normalize and store (token-major bf16) ----
#pragma unroll
  for (int r = 0; r < 8; ++r) {
    const float inv = 1.0f / lrow[r];
    const size_t row = (size_t)(qbase + r + 8 * hi) * 256;
#pragma unroll
    for (int fn = 0; fn < 4; ++fn)
      outB[row + hoff + fn * 16 + l16] = (__bf16)(o[fn][r] * inv);
  }
}

// ---------------------------------------------------------------------------
// LayerNorm: one 256-wide row per 256-thread block.  Optional residual add,
// optional second (unparameterized) LN on top.  Emits f32 + bf16 outputs.
// ---------------------------------------------------------------------------
__device__ __forceinline__ float block_sum256(float v) {
  __shared__ float sh[8];
  const int lane = threadIdx.x & 31, w = threadIdx.x >> 5;
#pragma unroll
  for (int m = 16; m > 0; m >>= 1) v += __shfl_xor(v, m, 32);
  __syncthreads();
  if (lane == 0) sh[w] = v;
  __syncthreads();
  float s = 0.0f;
#pragma unroll
  for (int i = 0; i < 8; ++i) s += sh[i];
  return s;
}

__global__ void __launch_bounds__(256)
ln_kernel(const float* __restrict__ x, const float* __restrict__ res,
          const float* __restrict__ g, const float* __restrict__ b,
          float* __restrict__ outF, __bf16* __restrict__ outB, int doubleLN) {
  const int row = blockIdx.x, tid = threadIdx.x;
  const size_t idx = (size_t)row * 256 + tid;
  float v = x[idx] + (res ? res[idx] : 0.0f);
  const float mu = block_sum256(v) * (1.0f / 256.0f);
  float c = v - mu;
  const float var = block_sum256(c * c) * (1.0f / 256.0f);
  float y = c * rsqrtf(var + 1e-5f) * g[tid] + b[tid];
  if (doubleLN) {
    const float mu2 = block_sum256(y) * (1.0f / 256.0f);
    const float c2 = y - mu2;
    const float var2 = block_sum256(c2 * c2) * (1.0f / 256.0f);
    y = c2 * rsqrtf(var2 + 1e-5f);
  }
  if (outF) outF[idx] = y;
  if (outB) outB[idx] = (__bf16)y;
}

// ---------------------------------------------------------------------------
// Small utility kernels
// ---------------------------------------------------------------------------
__global__ void cvt_bf16_kernel(const float* __restrict__ in,
                                __bf16* __restrict__ out, int n) {
  const int i = blockIdx.x * 256 + threadIdx.x;
  if (i < n) out[i] = (__bf16)in[i];
}

// [L][K][N] f32  ->  [L][N][K] bf16  (per-layer transpose for WMMA B frags)
__global__ void cvt_t_bf16_kernel(const float* __restrict__ in,
                                  __bf16* __restrict__ out, int L, int K, int N) {
  const int i = blockIdx.x * 256 + threadIdx.x;
  if (i >= L * K * N) return;
  const int l = i / (N * K);
  const int r = i - l * (N * K);
  const int n = r / K;
  const int k = r - n * K;
  out[i] = (__bf16)in[(size_t)l * K * N + (size_t)k * N + n];
}

__global__ void concat_kernel(const float* __restrict__ v,
                              const float* __restrict__ g,
                              const float* __restrict__ e,
                              float* __restrict__ X, __bf16* __restrict__ Xb) {
  const int i = blockIdx.x * 256 + threadIdx.x;  // over sz*256
  const int row = i >> 8, col = i & 255;
  float val;
  if (row < 2047)       val = v[i];
  else if (row == 2047) val = g[col];
  else                  val = e[i - 2048 * 256];
  X[i] = val;
  Xb[i] = (__bf16)val;
}

__global__ void reorder_out_kernel(const float* __restrict__ X,
                                   float* __restrict__ out, int sz, int n_v) {
  const int i = blockIdx.x * 256 + threadIdx.x;  // over sz*256
  const int row = i >> 8, col = i & 255;
  const int s = (row < n_v) ? row : ((row < sz - 1) ? row + 1 : n_v);
  out[i] = X[(size_t)s * 256 + col];
}

// ---------------------------------------------------------------------------
// Host orchestration
// ---------------------------------------------------------------------------
extern "C" void kernel_launch(void* const* d_in, const int* in_sizes, int n_in,
                              void* d_out, int out_size, void* d_ws, size_t ws_size,
                              hipStream_t stream) {
  const int sz = 4096, d = 256, dff = 1024, NLAYERS = 3, n_v = 2047;

  const float* v_in = (const float*)d_in[0];
  const float* e_in = (const float*)d_in[1];
  const float* g_in = (const float*)d_in[2];
  const int*   adj  = (const int*)d_in[3];
  const float* Wq = (const float*)d_in[4];  const float* bq = (const float*)d_in[5];
  const float* Wk = (const float*)d_in[6];  const float* bk = (const float*)d_in[7];
  const float* Wv = (const float*)d_in[8];  const float* bv = (const float*)d_in[9];
  const float* Wo = (const float*)d_in[10]; const float* bo = (const float*)d_in[11];
  const float* W1 = (const float*)d_in[12]; const float* b1 = (const float*)d_in[13];
  const float* W2 = (const float*)d_in[14]; const float* b2 = (const float*)d_in[15];
  const float* lng = (const float*)d_in[16];
  const float* lnb = (const float*)d_in[17];

  char* p = (char*)d_ws;
  auto alloc = [&](size_t bytes) -> void* {
    void* r = (void*)p;
    p += (bytes + 255) & ~(size_t)255;
    return r;
  };

  float*  X     = (float*)alloc(sizeof(float) * sz * d);
  __bf16* Xb    = (__bf16*)alloc(2ull * sz * d);
  __bf16* Qb    = (__bf16*)alloc(2ull * sz * d);
  __bf16* Kb    = (__bf16*)alloc(2ull * sz * d);
  __bf16* Vtb   = (__bf16*)alloc(2ull * sz * d);       // [feature][token]
  __bf16* attnb = (__bf16*)alloc(2ull * sz * d);
  float*  aof   = (float*)alloc(sizeof(float) * sz * d);
  float*  tf    = (float*)alloc(sizeof(float) * sz * d);
  __bf16* tb    = (__bf16*)alloc(2ull * sz * d);
  __bf16* h1b   = (__bf16*)alloc(2ull * sz * dff);
  float*  ffnf  = (float*)alloc(sizeof(float) * sz * d);
  // pre-transposed bf16 weights: [layer][N][K]
  __bf16* Wqb   = (__bf16*)alloc(2ull * NLAYERS * d * d);
  __bf16* Wkb   = (__bf16*)alloc(2ull * NLAYERS * d * d);
  __bf16* Wvb   = (__bf16*)alloc(2ull * NLAYERS * d * d);
  __bf16* Wob   = (__bf16*)alloc(2ull * NLAYERS * d * d);
  __bf16* W1b   = (__bf16*)alloc(2ull * NLAYERS * d * dff);
  __bf16* W2b   = (__bf16*)alloc(2ull * NLAYERS * dff * d);

  const int ndd = NLAYERS * d * d, ndf = NLAYERS * d * dff;
  cvt_t_bf16_kernel<<<(ndd + 255) / 256, 256, 0, stream>>>(Wq, Wqb, NLAYERS, d, d);
  cvt_t_bf16_kernel<<<(ndd + 255) / 256, 256, 0, stream>>>(Wk, Wkb, NLAYERS, d, d);
  cvt_t_bf16_kernel<<<(ndd + 255) / 256, 256, 0, stream>>>(Wv, Wvb, NLAYERS, d, d);
  cvt_t_bf16_kernel<<<(ndd + 255) / 256, 256, 0, stream>>>(Wo, Wob, NLAYERS, d, d);
  cvt_t_bf16_kernel<<<(ndf + 255) / 256, 256, 0, stream>>>(W1, W1b, NLAYERS, d, dff);
  cvt_t_bf16_kernel<<<(ndf + 255) / 256, 256, 0, stream>>>(W2, W2b, NLAYERS, dff, d);
  concat_kernel<<<(sz * d) / 256, 256, 0, stream>>>(v_in, g_in, e_in, X, Xb);

  const dim3 gP(d / GBN, sz / GBM);    // 256-wide GEMMs
  const dim3 gF(dff / GBN, sz / GBM);  // 1024-wide GEMM

  for (int i = 0; i < NLAYERS; ++i) {
    const __bf16* wq = Wqb + (size_t)i * d * d;
    const __bf16* wk = Wkb + (size_t)i * d * d;
    const __bf16* wv = Wvb + (size_t)i * d * d;
    const __bf16* wo = Wob + (size_t)i * d * d;
    const __bf16* w1 = W1b + (size_t)i * d * dff;   // [dff][d]
    const __bf16* w2 = W2b + (size_t)i * dff * d;   // [d][dff]

    gemm_bf16_kernel<<<gP, 256, 0, stream>>>(Xb, wq, bq + i * d, nullptr, Qb,
                                             nullptr, sz, d, d, 0);
    gemm_bf16_kernel<<<gP, 256, 0, stream>>>(Xb, wk, bk + i * d, nullptr, Kb,
                                             nullptr, sz, d, d, 0);
    gemm_bf16_kernel<<<gP, 256, 0, stream>>>(Xb, wv, bv + i * d, nullptr, nullptr,
                                             Vtb, sz, d, d, 0);
    attn_kernel<<<sz / 16, 128, 0, stream>>>(Qb, Kb, Vtb, adj, attnb, sz);
    gemm_bf16_kernel<<<gP, 256, 0, stream>>>(attnb, wo, bo + i * d, aof, nullptr,
                                             nullptr, sz, d, d, 0);
    ln_kernel<<<sz, 256, 0, stream>>>(aof, nullptr, lng + i * d, lnb + i * d,
                                      tf, tb, 0);
    gemm_bf16_kernel<<<gF, 256, 0, stream>>>(tb, w1, b1 + i * dff, nullptr, h1b,
                                             nullptr, sz, dff, d, 1);
    gemm_bf16_kernel<<<gP, 256, 0, stream>>>(h1b, w2, b2 + i * d, ffnf, nullptr,
                                             nullptr, sz, d, dff, 0);
    ln_kernel<<<sz, 256, 0, stream>>>(ffnf, tf, lng + i * d, lnb + i * d,
                                      X, Xb, 1);
  }

  reorder_out_kernel<<<(sz * d) / 256, 256, 0, stream>>>(X, (float*)d_out, sz, n_v);
  (void)in_sizes; (void)n_in; (void)out_size; (void)ws_size;
}
